// TextDecoder_66443144069180
// MI455X (gfx1250) — matrix operand order repaired and checked by reading, not compiled, not run on hardware
//
#include <hip/hip_runtime.h>

// ---------------- types for CDNA5 WMMA / TDM ----------------
typedef __attribute__((ext_vector_type(16))) __bf16 v16bf;
typedef __attribute__((ext_vector_type(8)))  __bf16 v8bf;
typedef __attribute__((ext_vector_type(8)))  float  v8f;
typedef __attribute__((ext_vector_type(4)))  unsigned int v4u;
typedef __attribute__((ext_vector_type(8)))  int v8i;
typedef __attribute__((ext_vector_type(4)))  int v4i;

#ifndef __has_builtin
#define __has_builtin(x) 0
#endif
#if __has_builtin(__builtin_amdgcn_tensor_load_to_lds) && __has_builtin(__builtin_amdgcn_s_wait_tensorcnt)
#define USE_TDM 1
#else
#define USE_TDM 0
#endif

#define BM 128
#define BN 64
#define BK 32

namespace {
constexpr int B_ = 4, T_ = 256, TA_ = 1500, D_ = 768, H_ = 12, L_ = 12;
constexpr int V_ = 51865, DH_ = 64, D4_ = 4 * D_;
constexpr long long BT_  = (long long)B_ * T_;    // 1024
constexpr long long BTA_ = (long long)B_ * TA_;   // 6000
}

__device__ __forceinline__ unsigned int f2bf(float f) {
  unsigned int u = __builtin_bit_cast(unsigned int, f);
  return (u + 0x7FFFu + ((u >> 16) & 1u)) >> 16;  // RNE, low 16 bits
}
__device__ __forceinline__ unsigned int pk2bf(float lo, float hi) {
  return f2bf(lo) | (f2bf(hi) << 16);
}

// ---------------- generic batched-strided GEMM (bf16 WMMA, f32 accum) ----------------
// A: [M,K] bf16 (aBf16=1) or fp32 (aBf16=0); B: [N,K] bf16 (contiguous k).
// batch offset decomposition: off = (z / Hmod)*s0 + (z % Hmod)*s1 (element units)
struct GemmP {
  const void* A; const unsigned short* B; const float* bias; const float* resid; float* C;
  int M, N, K, lda, ldb, ldc;
  long long sA0, sA1, sB0, sB1, sC0, sC1;
  int Hmod, aBf16, gelu; float scale;
};

__launch_bounds__(256)
__global__ void gemm_bf16_wmma(GemmP p) {
  __shared__ __align__(16) unsigned short As[2][BM * BK];  // [m][k]
  __shared__ __align__(16) unsigned short Bs[2][BN * BK];  // [n][k]
  const int z = blockIdx.z;
  const long long offA = (long long)(z / p.Hmod) * p.sA0 + (long long)(z % p.Hmod) * p.sA1;
  const long long offB = (long long)(z / p.Hmod) * p.sB0 + (long long)(z % p.Hmod) * p.sB1;
  const long long offC = (long long)(z / p.Hmod) * p.sC0 + (long long)(z % p.Hmod) * p.sC1;
  const float*          ApF = (const float*)p.A + offA;
  const unsigned short* ApH = (const unsigned short*)p.A + offA;
  const unsigned short* Bp  = p.B + offB;
  const int m0 = blockIdx.x * BM, n0 = blockIdx.y * BN;
  const int tid = threadIdx.x, lane = tid & 31, wave = tid >> 5;
  const int half = lane >> 4, l15 = lane & 15;

  v8f acc[4];
  #pragma unroll
  for (int nt = 0; nt < 4; ++nt)
    #pragma unroll
    for (int r = 0; r < 8; ++r) acc[nt][r] = 0.f;

  float aRegF[16];  // fp32-A staging (16 elem / thread / tile)
  uint4 aRegH[2];   // bf16-A staging (16 elem / thread / tile)

  // ---- A: global -> registers (branch-free; clamped addresses on edges) ----
  auto load_tile = [&](int k0) {
    const bool fullA = (m0 + BM <= p.M) && (k0 + BK <= p.K);
    if (p.aBf16) {
      if (fullA) {
        #pragma unroll
        for (int i = 0; i < 2; ++i) {            // 512 8-elem chunks / 256 thr
          int v8 = i * 256 + tid;
          int mm = v8 >> 2, kk = (v8 & 3) * 8;
          aRegH[i] = *reinterpret_cast<const uint4*>(
              ApH + (long long)(m0 + mm) * p.lda + (k0 + kk));
        }
      } else {
        unsigned short* ah = reinterpret_cast<unsigned short*>(aRegH);
        #pragma unroll
        for (int i = 0; i < 16; ++i) {
          int idx = i * 256 + tid;
          int mm = idx >> 5, kk = idx & (BK - 1);
          int gm = m0 + mm, gk = k0 + kk;
          int cm = gm < p.M ? gm : p.M - 1, ck = gk < p.K ? gk : p.K - 1;
          unsigned short v = ApH[(long long)cm * p.lda + ck];
          ah[i] = (gm < p.M && gk < p.K) ? v : (unsigned short)0;
        }
      }
    } else {
      if (fullA) {
        #pragma unroll
        for (int i = 0; i < 4; ++i) {            // 1024 float4 / 256 thr
          int v4 = i * 256 + tid;
          int mm = v4 >> 3, kk = (v4 & 7) * 4;
          const float4 f = *reinterpret_cast<const float4*>(
              ApF + (long long)(m0 + mm) * p.lda + (k0 + kk));
          aRegF[4 * i + 0] = f.x; aRegF[4 * i + 1] = f.y;
          aRegF[4 * i + 2] = f.z; aRegF[4 * i + 3] = f.w;
        }
      } else {
        #pragma unroll
        for (int i = 0; i < 16; ++i) {
          int idx = i * 256 + tid;
          int mm = idx >> 5, kk = idx & (BK - 1);
          int gm = m0 + mm, gk = k0 + kk;
          int cm = gm < p.M ? gm : p.M - 1, ck = gk < p.K ? gk : p.K - 1;
          float v = ApF[(long long)cm * p.lda + ck];
          aRegF[i] = (gm < p.M && gk < p.K) ? v : 0.f;
        }
      }
    }
  };

  // ---- A: registers -> LDS ----
  auto store_tile = [&](int buf, int k0) {
    const bool fullA = (m0 + BM <= p.M) && (k0 + BK <= p.K);
    if (p.aBf16) {
      if (fullA) {
        #pragma unroll
        for (int i = 0; i < 2; ++i) {
          int v8 = i * 256 + tid;
          int mm = v8 >> 2, kk = (v8 & 3) * 8;
          *reinterpret_cast<uint4*>(&As[buf][mm * BK + kk]) = aRegH[i];
        }
      } else {
        const unsigned short* ah = reinterpret_cast<const unsigned short*>(aRegH);
        #pragma unroll
        for (int i = 0; i < 16; ++i) {
          int idx = i * 256 + tid;
          int mm = idx >> 5, kk = idx & (BK - 1);
          As[buf][mm * BK + kk] = ah[i];
        }
      }
    } else {
      if (fullA) {
        #pragma unroll
        for (int i = 0; i < 4; ++i) {
          int v4 = i * 256 + tid;
          int mm = v4 >> 3, kk = (v4 & 7) * 4;
          uint2 u;
          u.x = pk2bf(aRegF[4 * i + 0], aRegF[4 * i + 1]);
          u.y = pk2bf(aRegF[4 * i + 2], aRegF[4 * i + 3]);
          *reinterpret_cast<uint2*>(&As[buf][mm * BK + kk]) = u;
        }
      } else {
        #pragma unroll
        for (int i = 0; i < 16; ++i) {
          int idx = i * 256 + tid;
          int mm = idx >> 5, kk = idx & (BK - 1);
          As[buf][mm * BK + kk] = (unsigned short)f2bf(aRegF[i]);
        }
      }
    }
  };

  // ---- B: staged by the Tensor Data Mover (async, OOB zero-filled) ----
#if USE_TDM
  auto tdm_issue = [&](int buf, int k0) {
    unsigned lds = (unsigned)(size_t)(&Bs[buf][0]);
    unsigned long long ga =
        (unsigned long long)(size_t)(Bp + (long long)n0 * p.ldb + k0);
    unsigned td0 = (unsigned)(p.K - k0);   // remaining K extent (elements)
    unsigned td1 = (unsigned)(p.N - n0);   // remaining N extent (rows)
    unsigned long long st0 = (unsigned long long)p.ldb;
    v4u g0;
    g0[0] = 1u;                                            // count=1
    g0[1] = lds;                                           // lds_addr
    g0[2] = (unsigned)(ga & 0xFFFFFFFFull);                // global_addr lo
    g0[3] = (unsigned)((ga >> 32) & 0x1FFFFFFull) | (2u << 30);  // hi | type=2
    v8i g1;
    g1[0] = (int)(1u << 16);                               // data_size=1 (2B)
    g1[1] = (int)((td0 & 0xFFFFu) << 16);                  // tensor_dim0[15:0]
    g1[2] = (int)((td0 >> 16) | ((td1 & 0xFFFFu) << 16));  // td0 hi | td1 lo
    g1[3] = (int)((td1 >> 16) | ((unsigned)BK << 16));     // td1 hi | tile_dim0=32
    g1[4] = (int)BN;                                       // tile_dim1=64, tile_dim2=0
    g1[5] = (int)(st0 & 0xFFFFFFFFull);                    // dim0 stride lo
    g1[6] = (int)((st0 >> 32) & 0xFFFFu);                  // dim0 stride hi
    g1[7] = 0;
    v4i z4 = {0, 0, 0, 0};
#if defined(__clang_major__) && __clang_major__ >= 23
    v8i z8 = {0, 0, 0, 0, 0, 0, 0, 0};
    __builtin_amdgcn_tensor_load_to_lds(g0, g1, z4, z4, z8, 0);
#else
    __builtin_amdgcn_tensor_load_to_lds(g0, g1, z4, z4, 0);
#endif
  };
#endif
  auto stage_b = [&](int buf, int k0) {
#if USE_TDM
    if (wave == 0) tdm_issue(buf, k0);
#else
    const bool fullB = (n0 + BN <= p.N) && (k0 + BK <= p.K);
    if (fullB) {
      int ch = tid;                                  // 256 8-elem chunks
      int nn = ch >> 2, kk = (ch & 3) * 8;
      *reinterpret_cast<uint4*>(&Bs[buf][nn * BK + kk]) =
          *reinterpret_cast<const uint4*>(Bp + (long long)(n0 + nn) * p.ldb + k0 + kk);
    } else {
      #pragma unroll
      for (int i = 0; i < 8; ++i) {
        int idx = i * 256 + tid;
        int nn = idx >> 5, kk = idx & (BK - 1);
        int gn = n0 + nn, gk = k0 + kk;
        int cn = gn < p.N ? gn : p.N - 1, ck = gk < p.K ? gk : p.K - 1;
        unsigned short v = Bp[(long long)cn * p.ldb + ck];
        Bs[buf][nn * BK + kk] = (gn < p.N && gk < p.K) ? v : (unsigned short)0;
      }
    }
#endif
  };

  // ---- WMMA compute on one staged tile ----
  auto compute = [&](int buf) {
    v8bf alo = *reinterpret_cast<const v8bf*>(&As[buf][(wave * 16 + l15) * BK + 8 * half]);
    v8bf ahi = *reinterpret_cast<const v8bf*>(&As[buf][(wave * 16 + l15) * BK + 16 + 8 * half]);
    v16bf afrag;
    #pragma unroll
    for (int e = 0; e < 8; ++e) { afrag[e] = alo[e]; afrag[8 + e] = ahi[e]; }
    #pragma unroll
    for (int nt = 0; nt < 4; ++nt) {
      const unsigned short* bb = &Bs[buf][(nt * 16 + l15) * BK + 8 * half];
      v8bf b0 = *reinterpret_cast<const v8bf*>(bb);
      v8bf b1 = *reinterpret_cast<const v8bf*>(bb + 16);
      v16bf bfrag;
      #pragma unroll
      for (int e = 0; e < 8; ++e) { bfrag[e] = b0[e]; bfrag[8 + e] = b1[e]; }
      acc[nt] = __builtin_amdgcn_wmma_f32_16x16x32_bf16(
          false, afrag, false, bfrag, (short)0, acc[nt], false, false);
    }
  };

  // ---- software-pipelined main loop: one barrier per k-step ----
  stage_b(0, 0);           // TDM for tile 0 in flight
  load_tile(0);
  store_tile(0, 0);
  int cur = 0;
  for (int k0 = 0; k0 < p.K; k0 += BK) {
    const bool hasNext = (k0 + BK) < p.K;
    if (hasNext) load_tile(k0 + BK);          // A(t+1) global loads in flight
#if USE_TDM
    if (wave == 0) __builtin_amdgcn_s_wait_tensorcnt(0);  // B(t) landed in LDS
#endif
    __syncthreads();                          // As[cur]+Bs[cur] visible to all
    if (hasNext) stage_b(cur ^ 1, k0 + BK);   // TDM B(t+1) overlaps compute(t)
    compute(cur);
    if (hasNext) store_tile(cur ^ 1, k0 + BK);
    cur ^= 1;
  }

  // ---- epilogue: C VGPR r -> M = r + 8*half ; N = l15 ----
  #pragma unroll
  for (int nt = 0; nt < 4; ++nt) {
    int gn = n0 + nt * 16 + l15;
    if (gn >= p.N) continue;
    float bv = p.bias ? p.bias[gn] : 0.f;
    #pragma unroll
    for (int r = 0; r < 8; ++r) {
      int gm = m0 + wave * 16 + 8 * half + r;
      if (gm >= p.M) continue;
      float v = (acc[nt][r] + bv) * p.scale;
      if (p.gelu) v = 0.5f * v * (1.f + erff(v * 0.70710678118f));
      if (p.resid) v += p.resid[(long long)gm * p.ldc + gn];
      p.C[offC + (long long)gm * p.ldc + gn] = v;
    }
  }
}

// ---------------- fp32 -> bf16 streaming convert (8 elem / thread) ----------------
__global__ void cvt_bf16_kernel(const float* __restrict__ in, unsigned short* __restrict__ out,
                                long long n8) {
  long long i = (long long)blockIdx.x * blockDim.x + threadIdx.x;
  if (i >= n8) return;
  const float4* p = reinterpret_cast<const float4*>(in + i * 8);
  float4 a = p[0], b = p[1];
  uint4 u;
  u.x = pk2bf(a.x, a.y); u.y = pk2bf(a.z, a.w);
  u.z = pk2bf(b.x, b.y); u.w = pk2bf(b.z, b.w);
  *reinterpret_cast<uint4*>(out + i * 8) = u;
}

// ---------------- fp32 [R,C] -> bf16 [C,R] transpose-convert (LDS tiled) ----------------
__global__ void cvtT_bf16_kernel(const float* __restrict__ in, unsigned short* __restrict__ out,
                                 int R, int C) {
  __shared__ float tile[32][33];
  int rb = blockIdx.y * 32, cb = blockIdx.x * 32;
  int tx = threadIdx.x & 31, ty = threadIdx.x >> 5;  // 32 x 8
  for (int j = ty; j < 32; j += 8) {
    int r = rb + j, c = cb + tx;
    tile[j][tx] = (r < R && c < C) ? in[(long long)r * C + c] : 0.f;
  }
  __syncthreads();
  for (int j = ty; j < 32; j += 8) {
    int c = cb + j, r = rb + tx;
    if (c < C && r < R) out[(long long)c * R + r] = (unsigned short)f2bf(tile[tx][j]);
  }
}

// ---------------- embedding gather + positional add ----------------
__global__ void embed_kernel(const int* __restrict__ tokens,
                             const float* __restrict__ tok_emb,
                             const float* __restrict__ pos_emb,
                             float* __restrict__ x) {
  long long i = (long long)blockIdx.x * blockDim.x + threadIdx.x;
  if (i >= BT_ * D_) return;
  int d = (int)(i % D_);
  long long bt = i / D_;
  int t = (int)(bt % T_);
  x[i] = tok_emb[(long long)tokens[bt] * D_ + d] + pos_emb[(long long)t * D_ + d];
}

// ---------------- LayerNorm: one wave32 per row ----------------
__global__ void layernorm_kernel(const float* __restrict__ x, const float* __restrict__ g,
                                 const float* __restrict__ b, float* __restrict__ out,
                                 int rows, int D) {
  int row = blockIdx.x * (blockDim.x >> 5) + (threadIdx.x >> 5);
  int lane = threadIdx.x & 31;
  if (row >= rows) return;
  const float* px = x + (long long)row * D;
  float s = 0.f, ss = 0.f;
  for (int i = lane; i < D; i += 32) { float v = px[i]; s += v; ss += v * v; }
  #pragma unroll
  for (int o = 16; o > 0; o >>= 1) { s += __shfl_xor(s, o, 32); ss += __shfl_xor(ss, o, 32); }
  float mean = s / D;
  float inv = rsqrtf(fmaxf(ss / D - mean * mean, 0.f) + 1e-5f);
  float* po = out + (long long)row * D;
  for (int i = lane; i < D; i += 32) po[i] = (px[i] - mean) * inv * g[i] + b[i];
}

// ---------------- softmax: one 256-thread block per row; optional causal ----------------
__global__ void softmax_kernel(float* __restrict__ x, int rows, int ncols, int rowmod, int causal) {
  __shared__ float red[256];
  const int row = blockIdx.x;
  if (row >= rows) return;
  const int tid = threadIdx.x;
  const int limit = causal ? (row % rowmod) + 1 : ncols;
  float* p = x + (long long)row * ncols;
  float mx = -3.4e38f;
  for (int i = tid; i < limit; i += 256) mx = fmaxf(mx, p[i]);
  red[tid] = mx; __syncthreads();
  for (int s = 128; s > 0; s >>= 1) { if (tid < s) red[tid] = fmaxf(red[tid], red[tid + s]); __syncthreads(); }
  mx = red[0]; __syncthreads();
  float sum = 0.f;
  for (int i = tid; i < limit; i += 256) { float e = __expf(p[i] - mx); p[i] = e; sum += e; }
  red[tid] = sum; __syncthreads();
  for (int s = 128; s > 0; s >>= 1) { if (tid < s) red[tid] += red[tid + s]; __syncthreads(); }
  float inv = 1.f / red[0];
  for (int i = tid; i < ncols; i += 256) p[i] = (i < limit) ? p[i] * inv : 0.f;
}

// ---------------- host driver ----------------
extern "C" void kernel_launch(void* const* d_in, const int* in_sizes, int n_in,
                              void* d_out, int out_size, void* d_ws, size_t ws_size,
                              hipStream_t stream) {
  (void)in_sizes; (void)n_in; (void)out_size; (void)ws_size;
  const int*   tokens  = (const int*)  d_in[0];
  const float* xa      = (const float*)d_in[1];
  const float* tok_emb = (const float*)d_in[2];
  const float* pos_emb = (const float*)d_in[3];
  const float* Wq  = (const float*)d_in[4];  const float* bq  = (const float*)d_in[5];
  const float* Wk  = (const float*)d_in[6];
  const float* Wv  = (const float*)d_in[7];  const float* bv  = (const float*)d_in[8];
  const float* Wo  = (const float*)d_in[9];  const float* bo  = (const float*)d_in[10];
  const float* ln1g = (const float*)d_in[11]; const float* ln1b = (const float*)d_in[12];
  const float* Wcq = (const float*)d_in[13]; const float* bcq = (const float*)d_in[14];
  const float* Wck = (const float*)d_in[15];
  const float* Wcv = (const float*)d_in[16]; const float* bcv = (const float*)d_in[17];
  const float* Wco = (const float*)d_in[18]; const float* bco = (const float*)d_in[19];
  const float* ln2g = (const float*)d_in[20]; const float* ln2b = (const float*)d_in[21];
  const float* W1 = (const float*)d_in[22]; const float* b1 = (const float*)d_in[23];
  const float* W2 = (const float*)d_in[24]; const float* b2 = (const float*)d_in[25];
  const float* ln3g = (const float*)d_in[26]; const float* ln3b = (const float*)d_in[27];
  const float* lnfg = (const float*)d_in[28]; const float* lnfb = (const float*)d_in[29];
  float* dout = (float*)d_out;

  // ---- workspace layout ----
  float* ws = (float*)d_ws;
  float* xbuf  = ws;                                    // [B*T, D]
  float* hbuf  = xbuf  + BT_ * D_;
  float* qbuf  = hbuf  + BT_ * D_;
  float* kbuf  = qbuf  + BT_ * D_;
  float* vbuf  = kbuf  + BT_ * D_;
  float* obuf  = vbuf  + BT_ * D_;
  float* sbuf  = obuf  + BT_ * D_;                      // [B,H,T,T]
  float* ckbuf = sbuf  + (long long)B_ * H_ * T_ * T_;  // [B*TA, D]
  float* cvbuf = ckbuf + BTA_ * D_;
  float* sxbuf = cvbuf + BTA_ * D_;                     // [B,H,T,TA]
  float* mbuf  = sxbuf + (long long)B_ * H_ * T_ * TA_; // [B*T, 4D]
  unsigned short* bfp  = (unsigned short*)(mbuf + BT_ * D4_);
  unsigned short* xabf = bfp;  bfp += BTA_ * D_;        // bf16 copies
  unsigned short* tokbf= bfp;  bfp += (long long)V_ * D_;
  unsigned short* hbf  = bfp;  bfp += BT_ * D_;
  unsigned short* qbf  = bfp;  bfp += BT_ * D_;
  unsigned short* kbf  = bfp;  bfp += BT_ * D_;
  unsigned short* obf  = bfp;  bfp += BT_ * D_;
  unsigned short* mbf  = bfp;  bfp += BT_ * D4_;
  unsigned short* ckbf = bfp;  bfp += BTA_ * D_;
  unsigned short* vT   = bfp;  bfp += BT_ * D_;         // [D, B*T]
  unsigned short* cvT  = bfp;  bfp += BTA_ * D_;        // [D, B*TA]
  unsigned short* wT   = bfp;                           // [N,K] weight scratch (<= 4D*D)

  const long long chw_per = (long long)H_ * T_ * TA_;
  const long long chw_off = BT_ * V_;
  const long long kv_off  = chw_off + (L_ / 2) * chw_per;
  const long long kvslot  = BT_ * D_;
  const long long DD = (long long)D_ * D_;

  auto gemm = [&](const void* A, int aBf16, const unsigned short* Bm, const float* bias,
                  const float* resid, float* C, int M, int N, int K, int lda, int ldb, int ldc,
                  long long sA0, long long sA1, long long sB0, long long sB1,
                  long long sC0, long long sC1, int Hmod, int batch, int gelu, float scale) {
    GemmP p{A, Bm, bias, resid, C, M, N, K, lda, ldb, ldc,
            sA0, sA1, sB0, sB1, sC0, sC1, Hmod, aBf16, gelu, scale};
    dim3 g((M + BM - 1) / BM, (N + BN - 1) / BN, batch);
    gemm_bf16_wmma<<<g, 256, 0, stream>>>(p);
  };
  auto cvt = [&](const float* in, unsigned short* out, long long n) {
    long long n8 = n / 8;
    cvt_bf16_kernel<<<dim3((unsigned)((n8 + 255) / 256)), 256, 0, stream>>>(in, out, n8);
  };
  auto cvtT = [&](const float* in, unsigned short* out, int R, int C) {
    cvtT_bf16_kernel<<<dim3((C + 31) / 32, (R + 31) / 32), 256, 0, stream>>>(in, out, R, C);
  };
  auto ln = [&](const float* in, const float* g_, const float* b_, float* out_, int rows) {
    layernorm_kernel<<<dim3((rows + 7) / 8), 256, 0, stream>>>(in, g_, b_, out_, rows, D_);
  };

  // once: embed + layer-invariant bf16 copies
  embed_kernel<<<dim3((unsigned)((BT_ * D_ + 255) / 256)), 256, 0, stream>>>(tokens, tok_emb, pos_emb, xbuf);
  cvt(xa, xabf, BTA_ * D_);
  cvt(tok_emb, tokbf, (long long)V_ * D_);

  for (int l = 0; l < L_; ++l) {
    // ---- masked self-attention ----
    ln(xbuf, ln1g + (long long)l * D_, ln1b + (long long)l * D_, hbuf, (int)BT_);
    cvt(hbuf, hbf, BT_ * D_);
    cvtT(Wq + l * DD, wT, D_, D_);
    gemm(hbf, 1, wT, bq + (long long)l * D_, nullptr, qbuf,
         (int)BT_, D_, D_, D_, D_, D_, 0, 0, 0, 0, 0, 0, 1, 1, 0, 0.125f);
    cvtT(Wk + l * DD, wT, D_, D_);
    gemm(hbf, 1, wT, nullptr, nullptr, kbuf,
         (int)BT_, D_, D_, D_, D_, D_, 0, 0, 0, 0, 0, 0, 1, 1, 0, 1.f);
    cvtT(Wv + l * DD, wT, D_, D_);
    gemm(hbf, 1, wT, bv + (long long)l * D_, nullptr, vbuf,
         (int)BT_, D_, D_, D_, D_, D_, 0, 0, 0, 0, 0, 0, 1, 1, 0, 1.f);
    hipMemcpyAsync(dout + kv_off + (long long)(2 * l) * kvslot, kbuf,
                   (size_t)kvslot * sizeof(float), hipMemcpyDeviceToDevice, stream);
    hipMemcpyAsync(dout + kv_off + (long long)(2 * l + 1) * kvslot, vbuf,
                   (size_t)kvslot * sizeof(float), hipMemcpyDeviceToDevice, stream);
    cvt(qbuf, qbf, BT_ * D_);
    cvt(kbuf, kbf, BT_ * D_);
    // scores[b,h] = q @ k^T
    gemm(qbf, 1, kbf, nullptr, nullptr, sbuf,
         T_, T_, DH_, D_, D_, T_,
         (long long)T_ * D_, DH_, (long long)T_ * D_, DH_,
         (long long)H_ * T_ * T_, (long long)T_ * T_,
         H_, B_ * H_, 0, 1.f);
    softmax_kernel<<<dim3(B_ * H_ * T_), 256, 0, stream>>>(sbuf, B_ * H_ * T_, T_, T_, 1);
    cvtT(vbuf, vT, (int)BT_, D_);                 // vT: [D, B*T]
    gemm(sbuf, 0, vT, nullptr, nullptr, obuf,
         T_, DH_, T_, T_, (int)BT_, D_,
         (long long)H_ * T_ * T_, (long long)T_ * T_, T_, (long long)DH_ * BT_,
         (long long)T_ * D_, DH_,
         H_, B_ * H_, 0, 1.f);
    cvt(obuf, obf, BT_ * D_);
    cvtT(Wo + l * DD, wT, D_, D_);
    gemm(obf, 1, wT, bo + (long long)l * D_, xbuf, xbuf,
         (int)BT_, D_, D_, D_, D_, D_, 0, 0, 0, 0, 0, 0, 1, 1, 0, 1.f);

    // ---- cross-attention ----
    ln(xbuf, ln2g + (long long)l * D_, ln2b + (long long)l * D_, hbuf, (int)BT_);
    cvt(hbuf, hbf, BT_ * D_);
    cvtT(Wcq + l * DD, wT, D_, D_);
    gemm(hbf, 1, wT, bcq + (long long)l * D_, nullptr, qbuf,
         (int)BT_, D_, D_, D_, D_, D_, 0, 0, 0, 0, 0, 0, 1, 1, 0, 0.125f);
    cvt(qbuf, qbf, BT_ * D_);
    cvtT(Wck + l * DD, wT, D_, D_);
    gemm(xabf, 1, wT, nullptr, nullptr, ckbuf,
         (int)BTA_, D_, D_, D_, D_, D_, 0, 0, 0, 0, 0, 0, 1, 1, 0, 1.f);
    cvt(ckbuf, ckbf, BTA_ * D_);
    cvtT(Wcv + l * DD, wT, D_, D_);
    gemm(xabf, 1, wT, bcv + (long long)l * D_, nullptr, cvbuf,
         (int)BTA_, D_, D_, D_, D_, D_, 0, 0, 0, 0, 0, 0, 1, 1, 0, 1.f);
    cvtT(cvbuf, cvT, (int)BTA_, D_);              // cvT: [D, B*TA]
    gemm(qbf, 1, ckbf, nullptr, nullptr, sxbuf,
         T_, TA_, DH_, D_, D_, TA_,
         (long long)T_ * D_, DH_, (long long)TA_ * D_, DH_,
         (long long)H_ * T_ * TA_, (long long)T_ * TA_,
         H_, B_ * H_, 0, 1.f);
    if (l >= L_ / 2)  // beam-0 pre-softmax cross scores
      hipMemcpyAsync(dout + chw_off + (long long)(l - L_ / 2) * chw_per, sxbuf,
                     (size_t)chw_per * sizeof(float), hipMemcpyDeviceToDevice, stream);
    softmax_kernel<<<dim3(B_ * H_ * T_), 256, 0, stream>>>(sxbuf, B_ * H_ * T_, TA_, T_, 0);
    gemm(sxbuf, 0, cvT, nullptr, nullptr, obuf,
         T_, DH_, TA_, TA_, (int)BTA_, D_,
         (long long)H_ * T_ * TA_, (long long)T_ * TA_, TA_, (long long)DH_ * BTA_,
         (long long)T_ * D_, DH_,
         H_, B_ * H_, 0, 1.f);
    cvt(obuf, obf, BT_ * D_);
    cvtT(Wco + l * DD, wT, D_, D_);
    gemm(obf, 1, wT, bco + (long long)l * D_, xbuf, xbuf,
         (int)BT_, D_, D_, D_, D_, D_, 0, 0, 0, 0, 0, 0, 1, 1, 0, 1.f);

    // ---- MLP ----
    ln(xbuf, ln3g + (long long)l * D_, ln3b + (long long)l * D_, hbuf, (int)BT_);
    cvt(hbuf, hbf, BT_ * D_);
    cvtT(W1 + (long long)l * D_ * D4_, wT, D_, D4_);     // [4D, D]
    gemm(hbf, 1, wT, b1 + (long long)l * D4_, nullptr, mbuf,
         (int)BT_, D4_, D_, D_, D_, D4_, 0, 0, 0, 0, 0, 0, 1, 1, 1, 1.f);
    cvt(mbuf, mbf, BT_ * D4_);
    cvtT(W2 + (long long)l * D4_ * D_, wT, D4_, D_);     // [D, 4D]
    gemm(mbf, 1, wT, b2 + (long long)l * D_, xbuf, xbuf,
         (int)BT_, D_, D4_, D4_, D4_, D_, 0, 0, 0, 0, 0, 0, 1, 1, 0, 1.f);
  }

  // final LN + tied-embedding logits (tokbf is [V, D] = [N, K])
  ln(xbuf, lnfg, lnfb, hbuf, (int)BT_);
  cvt(hbuf, hbf, BT_ * D_);
  gemm(hbf, 1, tokbf, nullptr, nullptr, dout,
       (int)BT_, V_, D_, D_, D_, V_, 0, 0, 0, 0, 0, 0, 1, 1, 0, 1.f);
}
